// RNNLayer_33646773797125
// MI455X (gfx1250) — matrix operand order repaired and checked by reading, not compiled, not run on hardware
//
#include <hip/hip_runtime.h>

// ---------------------------------------------------------------------------
// Leaky RNN: outputs[t] = h_t,  h_t = 0.9*h_{t-1} + 0.1*tanh(x_t W_ih^T + h_{t-1} W_hh^T + b)
// T=512, B=256, I=256, H=1024, fp32 in/out.
//
//  K0a/K0b: pack W_hh / W_ih into bf16 hi/lo B-fragments (WMMA lane layout)
//  K1:      P = X @ W_ih^T + bias -> d_out.  x tile converted ONCE per WG into
//           LDS as pre-swizzled A-fragments; K-loop = ds_load + WMMA only.
//  K2:      persistent 32-WG recurrence; h kept fp32 in registers (fixed tile
//           ownership) + bf16 hi/lo A-fragments in ws (ping-pong); software
//           grid barrier; P[t] overwritten in place with h_t.
// All GEMM math: v_wmma_f32_16x16x32_bf16, split-bf16 (hi+lo) ~fp32 accuracy.
// ---------------------------------------------------------------------------

typedef __attribute__((ext_vector_type(16))) __bf16 v16bf;
typedef __attribute__((ext_vector_type(8)))  float  v8f;

#define RNN_T 512
#define RNN_B 256
#define RNN_I 256
#define RNN_H 1024
#define DECAY 0.9f
#define ALPHA 0.1f
#define NWG_RNN 32
#define BH (RNN_B * RNN_H)            // 262144
#define WHH_N (RNN_H * RNN_H)         // 1048576 elems (hi part); lo at +WHH_N
#define WIH_N (RNN_H * RNN_I)         // 262144  elems (hi part); lo at +WIH_N
#define LDSA_N (4 * 8 * 512)          // 16384: 64 rows x 256 K as A-frags (hi); lo at +LDSA_N

#if __has_builtin(__builtin_amdgcn_tanhf)
#define FAST_TANH(x) __builtin_amdgcn_tanhf(x)
#elif __has_builtin(__builtin_amdgcn_tanh_f32)
#define FAST_TANH(x) __builtin_amdgcn_tanh_f32(x)
#else
#define FAST_TANH(x) tanhf(x)
#endif

static __device__ __forceinline__ void splitbf(float x, __bf16& hi, __bf16& lo) {
    hi = (__bf16)x;
    lo = (__bf16)(x - (float)hi);
}

static __device__ __forceinline__ v16bf ldfrag(const __bf16* p) {
    return *(const v16bf*)p;   // 32B aligned by construction
}

// split-bf16 fp32-accurate accumulate: acc += A*B (hi*hi + hi*lo + lo*hi)
static __device__ __forceinline__ v8f mma3(v8f acc, v16bf ah, v16bf al,
                                           v16bf bh, v16bf bl) {
    acc = __builtin_amdgcn_wmma_f32_16x16x32_bf16(false, ah, false, bh, (short)0, acc, false, false);
    acc = __builtin_amdgcn_wmma_f32_16x16x32_bf16(false, ah, false, bl, (short)0, acc, false, false);
    acc = __builtin_amdgcn_wmma_f32_16x16x32_bf16(false, al, false, bh, (short)0, acc, false, false);
    return acc;
}

// A-frag element decode: frag (mt, kt), within = lane'*16 + j
//   lane' = {hf=lane'>>4, r=lane'&15}; j<8 -> k=kt*32+8*hf+j ; j>=8 -> k=kt*32+16+8*hf+(j-8)
static __device__ __forceinline__ int afrag_k(int kt, int hf, int j) {
    return kt * 32 + (j < 8 ? 8 * hf + j : 16 + 8 * hf + (j - 8));
}

// ---------------------------------------------------------------------------
// K0: pack row-major W [N x Ktot] into B-fragments (B(k,n) = W[n,k]).
// Fragment (nt, kt): elem index = ((nt*KT + kt) << 9) + lane*16 + i
//   lane = {half=lane>>4, n=nt*16 + (lane&15)},  k = kt*32 + (i<8 ? 8*half+i : 16+8*half+(i-8))
// ---------------------------------------------------------------------------
__global__ void __launch_bounds__(256)
pack_bfrag_kernel(const float* __restrict__ W, int Ktot, int KT, int total,
                  __bf16* __restrict__ hiB, __bf16* __restrict__ loB)
{
    for (int e = blockIdx.x * 256 + threadIdx.x; e < total; e += gridDim.x * 256) {
        const int frag = e >> 9, w = e & 511;
        const int lane = w >> 4, i = w & 15;
        const int nt = frag / KT, kt = frag - nt * KT;
        const int n = nt * 16 + (lane & 15);
        const int k = kt * 32 + (i < 8 ? 8 * (lane >> 4) + i : 16 + 8 * (lane >> 4) + (i - 8));
        __bf16 h, l;
        splitbf(W[(size_t)n * Ktot + k], h, l);
        hiB[e] = h;
        loB[e] = l;
    }
}

// ---------------------------------------------------------------------------
// K1: P[m,n] = x[m,:] . W_ih[n,:] + bias[n]   (m in [0,T*B), n in [0,H))
// WG tile 64x128, 8 waves (2x4), wave tile 32x32.
// x tile staged in LDS as pre-swizzled hi/lo A-fragments (converted once/WG).
// ---------------------------------------------------------------------------
__global__ void __launch_bounds__(256)
rnn_proj_kernel(const float* __restrict__ x,      // [T*B, I]
                const __bf16* __restrict__ wihF,  // [WIH_N hi][WIH_N lo]
                const float* __restrict__ bias,   // [H]
                float* __restrict__ out)          // [T*B, H]
{
    __shared__ __bf16 ldsA[2 * LDSA_N];           // 64 KB: [hi 16384][lo 16384]

    const int lane = threadIdx.x & 31;
    const int wave = threadIdx.x >> 5;
    const int wr = wave >> 2, wc = wave & 3;
    const int mWG = blockIdx.x * 64;
    const int mBase = mWG + wr * 32;
    const int nBase = blockIdx.y * 128 + wc * 32;

    // ---- stage: convert WG's 64x256 x tile into LDS A-fragments (once) ----
    // frag (mt 0..3, kt 0..7): elem e = ((mt*8 + kt) << 9) + lane'*16 + j
    for (int e = threadIdx.x; e < LDSA_N; e += 256) {
        const int frag = e >> 9, w = e & 511;
        const int ln = w >> 4, j = w & 15;
        const int mt = frag >> 3, kt = frag & 7;
        const int hf = ln >> 4, r = ln & 15;
        const int k = afrag_k(kt, hf, j);
        __bf16 hh, ll;
        splitbf(x[(size_t)(mWG + mt * 16 + r) * RNN_I + k], hh, ll);
        ldsA[e] = hh;
        ldsA[e + LDSA_N] = ll;
    }
    __syncthreads();

    // ---- K loop: ds_load fragments + packed global B fragments + WMMA ----
    const __bf16* ap = ldsA + ((size_t)(wr * 2) * 8) * 512 + lane * 16;   // a=1 at +8*512
    const __bf16* bp = wihF + ((size_t)(nBase >> 4) * 8) * 512 + lane * 16; // b=1 at +8*512

    v8f acc[2][2] = {};
    for (int kt = 0; kt < 8; ++kt) {
        const v16bf a0h = ldfrag(ap),            a0l = ldfrag(ap + LDSA_N);
        const v16bf a1h = ldfrag(ap + 8 * 512),  a1l = ldfrag(ap + 8 * 512 + LDSA_N);
        const v16bf b0h = ldfrag(bp),            b0l = ldfrag(bp + WIH_N);
        const v16bf b1h = ldfrag(bp + 8 * 512),  b1l = ldfrag(bp + 8 * 512 + WIH_N);
        acc[0][0] = mma3(acc[0][0], a0h, a0l, b0h, b0l);
        acc[0][1] = mma3(acc[0][1], a0h, a0l, b1h, b1l);
        acc[1][0] = mma3(acc[1][0], a1h, a1l, b0h, b0l);
        acc[1][1] = mma3(acc[1][1], a1h, a1l, b1h, b1l);
        ap += 512;
        bp += 512;
    }

    const int half = lane >> 4, nIdx = lane & 15;
#pragma unroll
    for (int a = 0; a < 2; ++a)
#pragma unroll
        for (int b = 0; b < 2; ++b)
#pragma unroll
            for (int v = 0; v < 8; ++v) {
                const int m = mBase + a * 16 + v + 8 * half;
                const int n = nBase + b * 16 + nIdx;
                out[(size_t)m * RNN_H + n] = acc[a][b][v] + bias[n];
            }
}

// ---------------------------------------------------------------------------
// Grid-wide barrier: monotonic ticket/release counters in global memory.
// ---------------------------------------------------------------------------
static __device__ __forceinline__ void gridBarrier(unsigned* cnt, unsigned* release,
                                                   unsigned nwg) {
    __syncthreads();
    if (threadIdx.x == 0) {
        __threadfence();
        unsigned ticket = __hip_atomic_fetch_add(cnt, 1u, __ATOMIC_ACQ_REL,
                                                 __HIP_MEMORY_SCOPE_AGENT);
        unsigned gen = ticket / nwg;
        if (ticket % nwg == nwg - 1) {
            __hip_atomic_fetch_add(release, 1u, __ATOMIC_ACQ_REL,
                                   __HIP_MEMORY_SCOPE_AGENT);
        } else {
            while (__hip_atomic_load(release, __ATOMIC_ACQUIRE,
                                     __HIP_MEMORY_SCOPE_AGENT) <= gen) {
                __builtin_amdgcn_s_sleep(2);
            }
        }
        __threadfence();
    }
    __syncthreads();
}

// ---------------------------------------------------------------------------
// K2: persistent recurrence.  32 WGs = (B/64) x (H/128); wave tile 32x32.
// ---------------------------------------------------------------------------
__global__ void __launch_bounds__(256)
rnn_recur_kernel(const float* __restrict__ h0,    // [B, H]
                 const __bf16* __restrict__ whhF, // [WHH_N hi][WHH_N lo]
                 float* __restrict__ out,         // [T*B*H] outputs, then [B*H] h_last
                 __bf16* __restrict__ hA,         // [2][ BH hi | BH lo ]
                 unsigned* cnt, unsigned* release)
{
    const int lane  = threadIdx.x & 31;
    const int wave  = threadIdx.x >> 5;
    const int mBase = (blockIdx.x >> 3) * 64 + (wave >> 2) * 32;
    const int nBase = (blockIdx.x & 7) * 128 + (wave & 3) * 32;
    const int halfC = lane >> 4, nIdx = lane & 15;
    const int mt0 = mBase >> 4;          // first A row-tile index
    const int nt0 = nBase >> 4;          // first B col-tile index
    const int ktE = nBase >> 5;          // K-tile of the A-frags this wave produces
    const int halfW = nIdx >> 3, j0 = nIdx & 7;

    // ---- stage h_init into A-frag buffer 0 (hi/lo) ----
    for (int e = blockIdx.x * 256 + threadIdx.x; e < BH; e += NWG_RNN * 256) {
        const int frag = e >> 9, w = e & 511;
        const int ln = w >> 4, j = w & 15;
        const int mt = frag >> 5, kt = frag & 31;
        const int hf = ln >> 4, r = ln & 15;
        __bf16 hh, ll;
        splitbf(h0[(size_t)(mt * 16 + r) * RNN_H + afrag_k(kt, hf, j)], hh, ll);
        hA[e] = hh;
        hA[e + BH] = ll;
    }

    // ---- own fp32 h tile lives in registers across all steps ----
    float hreg[2][2][8];
#pragma unroll
    for (int a = 0; a < 2; ++a)
#pragma unroll
        for (int b = 0; b < 2; ++b)
#pragma unroll
            for (int v = 0; v < 8; ++v)
                hreg[a][b][v] =
                    h0[(size_t)(mBase + a * 16 + v + 8 * halfC) * RNN_H +
                       nBase + b * 16 + nIdx];

    gridBarrier(cnt, release, NWG_RNN);

    for (int t = 0; t < RNN_T; ++t) {
        const int rp = t & 1, wp = rp ^ 1;

        const __bf16* ap = hA + (size_t)rp * 2 * BH + ((size_t)mt0 * 32) * 512 + lane * 16;
        const __bf16* bp = whhF + ((size_t)nt0 * 32) * 512 + lane * 16;

        v8f acc[2][2] = {};
        for (int kt = 0; kt < 32; ++kt) {
            const v16bf a0h = ldfrag(ap),             a0l = ldfrag(ap + BH);
            const v16bf a1h = ldfrag(ap + 32 * 512),  a1l = ldfrag(ap + 32 * 512 + BH);
            const v16bf b0h = ldfrag(bp),             b0l = ldfrag(bp + WHH_N);
            const v16bf b1h = ldfrag(bp + 32 * 512),  b1l = ldfrag(bp + 32 * 512 + WHH_N);
            acc[0][0] = mma3(acc[0][0], a0h, a0l, b0h, b0l);
            acc[0][1] = mma3(acc[0][1], a0h, a0l, b1h, b1l);
            acc[1][0] = mma3(acc[1][0], a1h, a1l, b0h, b0l);
            acc[1][1] = mma3(acc[1][1], a1h, a1l, b1h, b1l);
            ap += 512;
            bp += 512;
        }

        // ---- epilogue: h_t = 0.9*h + 0.1*tanh(P + R); publish fp32 + bf16 frags
        float* outT = out + (size_t)t * BH;            // holds P[t]; becomes h_t
        __bf16* wf = hA + (size_t)wp * 2 * BH;         // write frag buffer (hi); lo at +BH
#pragma unroll
        for (int a = 0; a < 2; ++a)
#pragma unroll
            for (int b = 0; b < 2; ++b)
#pragma unroll
                for (int v = 0; v < 8; ++v) {
                    const int m = mBase + a * 16 + v + 8 * halfC;
                    const int n = nBase + b * 16 + nIdx;
                    const size_t idx = (size_t)m * RNN_H + n;
                    const float act = FAST_TANH(outT[idx] + acc[a][b][v]);
                    const float hn = DECAY * hreg[a][b][v] + ALPHA * act;
                    hreg[a][b][v] = hn;
                    outT[idx] = hn;
                    const int dest = (((mt0 + a) * 32 + ktE) << 9) +
                                     (halfW * 16 + v + 8 * halfC) * 16 +
                                     j0 + (b ? 8 : 0);
                    __bf16 hh = (__bf16)hn;
                    wf[dest] = hh;
                    wf[dest + BH] = (__bf16)(hn - (float)hh);
                }
        gridBarrier(cnt, release, NWG_RNN);
    }

    // ---- h_last straight from registers to tail of d_out ----
    float* tail = out + (size_t)RNN_T * BH;
#pragma unroll
    for (int a = 0; a < 2; ++a)
#pragma unroll
        for (int b = 0; b < 2; ++b)
#pragma unroll
            for (int v = 0; v < 8; ++v)
                tail[(size_t)(mBase + a * 16 + v + 8 * halfC) * RNN_H +
                     nBase + b * 16 + nIdx] = hreg[a][b][v];
}

// ---------------------------------------------------------------------------
extern "C" void kernel_launch(void* const* d_in, const int* in_sizes, int n_in,
                              void* d_out, int out_size, void* d_ws, size_t ws_size,
                              hipStream_t stream) {
    (void)in_sizes; (void)n_in; (void)out_size; (void)ws_size;
    const float* x    = (const float*)d_in[0];  // [T, B, I]
    const float* h0   = (const float*)d_in[1];  // [1, B, H]
    const float* Wih  = (const float*)d_in[2];  // [H, I]
    const float* Whh  = (const float*)d_in[3];  // [H, H]
    const float* bias = (const float*)d_in[4];  // [H]
    float* out = (float*)d_out;

    // workspace layout
    char* p = (char*)d_ws;
    unsigned* cnt     = (unsigned*)p;
    unsigned* release = cnt + 64;
    p += 4096;
    __bf16* whhF = (__bf16*)p;  p += (size_t)2 * WHH_N * sizeof(__bf16);  // 4 MB
    __bf16* wihF = (__bf16*)p;  p += (size_t)2 * WIH_N * sizeof(__bf16);  // 1 MB
    __bf16* hA   = (__bf16*)p;  p += (size_t)4 * BH * sizeof(__bf16);     // 2 MB

    hipMemsetAsync(d_ws, 0, 4096, stream);  // reset barrier state (deterministic)

    // K0: pack weights into B-fragments (hi/lo)
    pack_bfrag_kernel<<<dim3(WHH_N / 256), 256, 0, stream>>>(
        Whh, RNN_H, RNN_H / 32, WHH_N, whhF, whhF + WHH_N);
    pack_bfrag_kernel<<<dim3(WIH_N / 256), 256, 0, stream>>>(
        Wih, RNN_I, RNN_I / 32, WIH_N, wihF, wihF + WIH_N);

    // K1: P = X @ W_ih^T + bias  ->  d_out
    rnn_proj_kernel<<<dim3((RNN_T * RNN_B) / 64, RNN_H / 128), 256, 0, stream>>>(
        x, wihF, bias, out);

    // K2: persistent recurrence over 512 steps
    rnn_recur_kernel<<<dim3(NWG_RNN), 256, 0, stream>>>(
        h0, whhF, out, hA, cnt, release);
}
